// TemporalStructureAttention_17781164605593
// MI455X (gfx1250) — matrix operand order repaired and checked by reading, not compiled, not run on hardware
//
#include <hip/hip_runtime.h>
#include <hip/hip_bf16.h>

// Problem constants (match reference)
#define B_   8
#define S_   2048
#define D_   256      // d_model == d_in
#define NV_  8
#define F_   2048     // NUM_VAR * D_MODEL
#define NEGI -1e9f

typedef __attribute__((ext_vector_type(8)))  float  v8f;
typedef __attribute__((ext_vector_type(16))) __bf16 v16bf;

union V16 {
  v16bf v;
  __bf16 h[16];
  unsigned short u[16];
  uint4 q[2];
};

// ---------------------------------------------------------------------------
// Prep kernel A: Wt(bf16)[n][k] = W(f32)[k][n]   (256x256)
// grid = 256, block = 256
// ---------------------------------------------------------------------------
__global__ __launch_bounds__(256) void convw_kernel(
    const float* __restrict__ W, __bf16* __restrict__ Wt) {
  const int idx = blockIdx.x * 256 + threadIdx.x;
  const int n = idx >> 8;
  const int k = idx & 255;
  Wt[idx] = (__bf16)W[(k << 8) + n];
}

// ---------------------------------------------------------------------------
// Prep kernel B: vt(bf16)[b][n][k] = v(f32)[b][k][n]  via 32x32 LDS tiles
// grid = (S/32, F/32, B), block = 256
// ---------------------------------------------------------------------------
__global__ __launch_bounds__(256) void convv_kernel(
    const float* __restrict__ v, __bf16* __restrict__ vt) {
  __shared__ float t[32][33];
  const int tx = threadIdx.x & 31;
  const int ty = threadIdx.x >> 5;           // 0..7
  const int kt = blockIdx.x, nt = blockIdx.y, b = blockIdx.z;
#pragma unroll
  for (int it = 0; it < 4; ++it) {
    const int kk = ty + 8 * it;
    t[kk][tx] = v[((size_t)b * S_ + kt * 32 + kk) * F_ + nt * 32 + tx];
  }
  __syncthreads();
#pragma unroll
  for (int it = 0; it < 4; ++it) {
    const int nn = ty + 8 * it;
    vt[((size_t)b * F_ + nt * 32 + nn) * S_ + kt * 32 + tx] = (__bf16)t[tx][nn];
  }
}

// ---------------------------------------------------------------------------
// Kernel 1: Y(bf16)[R,256] = X(f32)[R,256] @ W + bias,  W given as Wt[n][k] bf16
// grid = (16, R/16), block = 32 (one wave per 16x16 tile)
// ---------------------------------------------------------------------------
__global__ __launch_bounds__(32) void proj_kernel(
    const float* __restrict__ X, const __bf16* __restrict__ Wt,
    const float* __restrict__ bias, __bf16* __restrict__ Y) {
  const int lane = threadIdx.x & 31;
  const int lb   = lane & 15;
  const int half = lane >> 4;
  const int mt = blockIdx.y, nt = blockIdx.x;
  const int m = mt * 16 + lb;
  const int n = nt * 16 + lb;
  const float*  arow = X + (size_t)m * D_;
  const __bf16* brow = Wt + (size_t)n * D_;
  v8f acc = {};
  for (int k0 = 0; k0 < D_; k0 += 32) {
    V16 A, Bf;
    // A: fp32 source -> bf16; two contiguous groups of 8 floats per half
    const float4 f0 = *(const float4*)(arow + k0 + 8 * half);
    const float4 f1 = *(const float4*)(arow + k0 + 8 * half + 4);
    const float4 f2 = *(const float4*)(arow + k0 + 16 + 8 * half);
    const float4 f3 = *(const float4*)(arow + k0 + 16 + 8 * half + 4);
    A.h[0] = (__bf16)f0.x; A.h[1] = (__bf16)f0.y; A.h[2] = (__bf16)f0.z; A.h[3] = (__bf16)f0.w;
    A.h[4] = (__bf16)f1.x; A.h[5] = (__bf16)f1.y; A.h[6] = (__bf16)f1.z; A.h[7] = (__bf16)f1.w;
    A.h[8]  = (__bf16)f2.x; A.h[9]  = (__bf16)f2.y; A.h[10] = (__bf16)f2.z; A.h[11] = (__bf16)f2.w;
    A.h[12] = (__bf16)f3.x; A.h[13] = (__bf16)f3.y; A.h[14] = (__bf16)f3.z; A.h[15] = (__bf16)f3.w;
    // B: bf16 transposed weights, two contiguous groups of 8 halves per half
    Bf.q[0] = *(const uint4*)(brow + k0 + 8 * half);
    Bf.q[1] = *(const uint4*)(brow + k0 + 16 + 8 * half);
    acc = __builtin_amdgcn_wmma_f32_16x16x32_bf16(false, A.v, false, Bf.v,
                                                  (short)0, acc, false, false);
  }
  const float bn = bias[n];
#pragma unroll
  for (int r = 0; r < 8; ++r) {
    const int row = mt * 16 + r + 8 * half;
    Y[(size_t)row * D_ + n] = (__bf16)(acc[r] + bn);
  }
}

// ---------------------------------------------------------------------------
// Kernel 2: causal scores + softmax. One workgroup = (batch, 16 query rows).
// 4 waves; scores strip [16][S] staged in dynamic LDS (128KB).
// Writes fp32 attn to d_out AND a bf16 copy to ws for the AV GEMM.
// ---------------------------------------------------------------------------
__global__ __launch_bounds__(128) void attn_kernel(
    const __bf16* __restrict__ qb16, const __bf16* __restrict__ kb16,
    float* __restrict__ attn, __bf16* __restrict__ at) {
  extern __shared__ float smem[];
  float* sc     = smem;              // [16][S_]
  float* redmax = smem + 16 * S_;    // [128]
  float* redsum = redmax + 128;      // [128]

  const int b  = blockIdx.x / (S_ / 16);
  const int qt = blockIdx.x % (S_ / 16);
  const int qb = qt * 16;
  const int tid  = threadIdx.x;
  const int wave = tid >> 5;
  const int lane = tid & 31;
  const int lb   = lane & 15;
  const int half = lane >> 4;
  const int qmax = qb + 15;
  const float scale = 0.0625f;  // 1/sqrt(256)

  const __bf16* qrow = qb16 + (size_t)(b * S_ + qb + lb) * D_;
  float*  attn_rowbase = attn + ((size_t)b * S_ + qb) * S_;
  __bf16* at_rowbase   = at   + ((size_t)b * S_ + qb) * S_;

  // Phase 1: score tiles via WMMA; fully-masked tiles -> direct zero store.
  for (int kt = wave; kt < S_ / 16; kt += 4) {
    const int kbase = kt * 16;
    if (kbase > qmax) {  // wave-uniform branch: EXEC stays all-ones
#pragma unroll
      for (int r = 0; r < 8; ++r) {
        const size_t off = (size_t)(r + 8 * half) * S_ + kbase + lb;
        attn_rowbase[off] = 0.0f;
        at_rowbase[off]   = (__bf16)0.0f;
      }
      continue;
    }
    const __bf16* krow = kb16 + (size_t)(b * S_ + kbase + lb) * D_;
    v8f acc = {};
#pragma unroll
    for (int k0 = 0; k0 < D_; k0 += 32) {
      V16 A, Bf;
      A.q[0]  = *(const uint4*)(qrow + k0 + 8 * half);
      A.q[1]  = *(const uint4*)(qrow + k0 + 16 + 8 * half);
      Bf.q[0] = *(const uint4*)(krow + k0 + 8 * half);
      Bf.q[1] = *(const uint4*)(krow + k0 + 16 + 8 * half);
      acc = __builtin_amdgcn_wmma_f32_16x16x32_bf16(false, A.v, false, Bf.v,
                                                    (short)0, acc, false, false);
    }
#pragma unroll
    for (int r = 0; r < 8; ++r) {
      const int row = r + 8 * half;
      float s = acc[r] * scale;
      if (kbase + lb > qb + row) s = NEGI;   // causal mask within tile
      sc[row * S_ + kbase + lb] = s;
    }
  }
  __syncthreads();

  // Phases 2-4: row-wise softmax over the active prefix [0, qb+16)
  const int row = tid >> 3;
  const int chunk = tid & 7;
  const int ncols = qb + 16;
  const int c0 = chunk * 256;
  int c1 = c0 + 256; if (c1 > ncols) c1 = ncols;

  float lmax = -3.0e38f;
  for (int c = c0; c < c1; ++c) lmax = fmaxf(lmax, sc[row * S_ + c]);
  redmax[tid] = lmax;
  __syncthreads();

  float rmax = -3.0e38f;
#pragma unroll
  for (int j = 0; j < 8; ++j) rmax = fmaxf(rmax, redmax[(row << 3) + j]);

  float lsum = 0.0f;
  for (int c = c0; c < c1; ++c) {
    const float e = __expf(sc[row * S_ + c] - rmax);
    sc[row * S_ + c] = e;
    lsum += e;
  }
  redsum[tid] = lsum;
  __syncthreads();

  float rsum = 0.0f;
#pragma unroll
  for (int j = 0; j < 8; ++j) rsum += redsum[(row << 3) + j];
  const float rinv = 1.0f / rsum;
  float*  arow2 = attn + ((size_t)b * S_ + qb + row) * S_;
  __bf16* atrow = at   + ((size_t)b * S_ + qb + row) * S_;
  for (int c = c0; c < c1; ++c) {
    const float p = sc[row * S_ + c] * rinv;
    arow2[c] = p;
    atrow[c] = (__bf16)p;
  }
}

// ---------------------------------------------------------------------------
// Kernel 3: out_(bf16)[B,S,F] = attn(bf16) @ v_(bf16, pre-transposed)
// 4x4 register-blocked: one wave computes a 64x64 output block.
// Inner loop: 8 frag loads (16x b128) : 16 wmma per 32-wide K step.
// K-loop truncated at causal boundary.
// grid = (F/64, S/64, B), block = 32
// ---------------------------------------------------------------------------
__global__ __launch_bounds__(32) void av_kernel(
    const __bf16* __restrict__ at, const __bf16* __restrict__ vt,
    __bf16* __restrict__ ob) {
  const int lane = threadIdx.x & 31;
  const int lb   = lane & 15;
  const int half = lane >> 4;
  const int nb = blockIdx.x * 64;
  const int qb = blockIdx.y * 64;
  const int b  = blockIdx.z;

  const __bf16* arow[4];
  const __bf16* brow[4];
#pragma unroll
  for (int qi = 0; qi < 4; ++qi)
    arow[qi] = at + ((size_t)b * S_ + qb + 16 * qi + lb) * S_;
#pragma unroll
  for (int ni = 0; ni < 4; ++ni)
    brow[ni] = vt + ((size_t)b * F_ + nb + 16 * ni + lb) * S_;

  v8f acc[4][4] = {};
  const int kend = qb + 64;
  for (int k0 = 0; k0 < kend; k0 += 32) {
    V16 A[4], Bv[4];
#pragma unroll
    for (int qi = 0; qi < 4; ++qi) {
      A[qi].q[0] = *(const uint4*)(arow[qi] + k0 + 8 * half);
      A[qi].q[1] = *(const uint4*)(arow[qi] + k0 + 16 + 8 * half);
    }
#pragma unroll
    for (int ni = 0; ni < 4; ++ni) {
      Bv[ni].q[0] = *(const uint4*)(brow[ni] + k0 + 8 * half);
      Bv[ni].q[1] = *(const uint4*)(brow[ni] + k0 + 16 + 8 * half);
    }
#pragma unroll
    for (int qi = 0; qi < 4; ++qi)
#pragma unroll
      for (int ni = 0; ni < 4; ++ni)
        acc[qi][ni] = __builtin_amdgcn_wmma_f32_16x16x32_bf16(
            false, A[qi].v, false, Bv[ni].v, (short)0, acc[qi][ni], false, false);
  }
#pragma unroll
  for (int qi = 0; qi < 4; ++qi)
#pragma unroll
    for (int ni = 0; ni < 4; ++ni)
#pragma unroll
      for (int r = 0; r < 8; ++r) {
        const int row = qb + 16 * qi + r + 8 * half;
        ob[((size_t)b * S_ + row) * F_ + nb + 16 * ni + lb] = (__bf16)acc[qi][ni][r];
      }
}

// ---------------------------------------------------------------------------
// Kernel 4: output(f32)[R2=B*S*NV, 256] = ob(bf16)[R2,256] @ Wl + bl
// (the [B,S,NV,256] view is row-contiguous -> flat GEMM; Wl given as Wlt[n][k])
// grid = (16, R2/16), block = 32
// ---------------------------------------------------------------------------
__global__ __launch_bounds__(32) void final_kernel(
    const __bf16* __restrict__ ob, const __bf16* __restrict__ Wlt,
    const float* __restrict__ bl, float* __restrict__ out) {
  const int lane = threadIdx.x & 31;
  const int lb   = lane & 15;
  const int half = lane >> 4;
  const int mt = blockIdx.y, nt = blockIdx.x;
  const int m = mt * 16 + lb;
  const int n = nt * 16 + lb;
  const __bf16* arow = ob  + (size_t)m * D_;
  const __bf16* brow = Wlt + (size_t)n * D_;
  v8f acc = {};
  for (int k0 = 0; k0 < D_; k0 += 32) {
    V16 A, Bf;
    A.q[0]  = *(const uint4*)(arow + k0 + 8 * half);
    A.q[1]  = *(const uint4*)(arow + k0 + 16 + 8 * half);
    Bf.q[0] = *(const uint4*)(brow + k0 + 8 * half);
    Bf.q[1] = *(const uint4*)(brow + k0 + 16 + 8 * half);
    acc = __builtin_amdgcn_wmma_f32_16x16x32_bf16(false, A.v, false, Bf.v,
                                                  (short)0, acc, false, false);
  }
  const float bn = bl[n];
#pragma unroll
  for (int r = 0; r < 8; ++r) {
    const int row = mt * 16 + r + 8 * half;
    out[(size_t)row * D_ + n] = acc[r] + bn;
  }
}

// ---------------------------------------------------------------------------
extern "C" void kernel_launch(void* const* d_in, const int* in_sizes, int n_in,
                              void* d_out, int out_size, void* d_ws, size_t ws_size,
                              hipStream_t stream) {
  const float* q  = (const float*)d_in[0];
  const float* k  = (const float*)d_in[1];
  const float* v  = (const float*)d_in[2];
  const float* Wq = (const float*)d_in[3];
  const float* bq = (const float*)d_in[4];
  const float* Wk = (const float*)d_in[5];
  const float* bk = (const float*)d_in[6];
  const float* Wl = (const float*)d_in[7];
  const float* bl = (const float*)d_in[8];

  float* out  = (float*)d_out;                               // [B,S,NV,D]
  float* attn = out + (size_t)B_ * S_ * NV_ * D_;            // [B,S,S]

  // workspace partition (bf16 elements)
  __bf16* qb16 = (__bf16*)d_ws;                              //  [B*S*D]
  __bf16* kb16 = qb16 + (size_t)B_ * S_ * D_;                //  [B*S*D]
  __bf16* ob16 = kb16 + (size_t)B_ * S_ * D_;                //  [B*S*F]
  __bf16* vt16 = ob16 + (size_t)B_ * S_ * F_;                //  [B*F*S] (v transposed)
  __bf16* at16 = vt16 + (size_t)B_ * F_ * S_;                //  [B*S*S] (attn bf16)
  __bf16* wqt  = at16 + (size_t)B_ * S_ * S_;                //  [D*D]
  __bf16* wkt  = wqt + (size_t)D_ * D_;
  __bf16* wlt  = wkt + (size_t)D_ * D_;

  // Prep: weight convert+transpose, v convert+transpose
  convw_kernel<<<dim3(D_ * D_ / 256), 256, 0, stream>>>(Wq, wqt);
  convw_kernel<<<dim3(D_ * D_ / 256), 256, 0, stream>>>(Wk, wkt);
  convw_kernel<<<dim3(D_ * D_ / 256), 256, 0, stream>>>(Wl, wlt);
  convv_kernel<<<dim3(S_ / 32, F_ / 32, B_), 256, 0, stream>>>(v, vt16);

  // Projections
  proj_kernel<<<dim3(D_ / 16, B_ * S_ / 16), 32, 0, stream>>>(q, wqt, bq, qb16);
  proj_kernel<<<dim3(D_ / 16, B_ * S_ / 16), 32, 0, stream>>>(k, wkt, bk, kb16);

  // Scores + softmax (fp32 attn out + bf16 attn staging)
  const size_t smem = (size_t)16 * S_ * sizeof(float) + 256 * sizeof(float);
  attn_kernel<<<dim3(B_ * S_ / 16), 128, smem, stream>>>(qb16, kb16, attn, at16);

  // attn @ V (register-blocked)
  av_kernel<<<dim3(F_ / 64, S_ / 64, B_), 32, 0, stream>>>(at16, vt16, ob16);

  // Final projection
  final_kernel<<<dim3(D_ / 16, B_ * S_ * NV_ / 16), 32, 0, stream>>>(ob16, wlt, bl, out);
}
